// Attention_81423989997987
// MI455X (gfx1250) — matrix operand order repaired and checked by reading, not compiled, not run on hardware
//
#include <hip/hip_runtime.h>

#define DIM    768
#define NSEQ   1024
#define BATCH  8
#define HEADS  12
#define HD     64
#define SCALE_F 0.125f   // 1/sqrt(64)

typedef __attribute__((ext_vector_type(16))) __bf16 v16bf;
typedef __attribute__((ext_vector_type(8)))  float  v8f;
typedef __attribute__((__vector_size__(16))) int    v4i_vs;   // matches builtin sig

// ---------- CDNA5 async global->LDS path (probe-guarded) ---------------------

#if __has_builtin(__builtin_amdgcn_global_load_async_to_lds_b128)
#define HAVE_ASYNC 1
#define ASYNC_COPY_B128(dst_lds, src_glb)                                      \
  __builtin_amdgcn_global_load_async_to_lds_b128(                              \
      (__attribute__((address_space(1))) v4i_vs*)(src_glb),                    \
      (__attribute__((address_space(3))) v4i_vs*)(dst_lds), 0, 0)
#else
#define HAVE_ASYNC 0
#endif

#if __has_builtin(__builtin_amdgcn_s_wait_asynccnt)
#define ASYNC_WAIT() __builtin_amdgcn_s_wait_asynccnt(0)
#elif HAVE_ASYNC
#define ASYNC_WAIT() asm volatile("s_wait_asynccnt 0" ::: "memory")
#else
#define ASYNC_WAIT()
#endif

// ---------- helpers ----------------------------------------------------------

__device__ __forceinline__ unsigned short f2bf(float f) {
  return __builtin_bit_cast(unsigned short, (__bf16)f);   // hw cvt op
}

__device__ __forceinline__ v8f zero_v8() {
  v8f z;
#pragma unroll
  for (int i = 0; i < 8; ++i) z[i] = 0.0f;
  return z;
}

__device__ __forceinline__ v8f wmma_bf16(v16bf a, v16bf b, v8f c) {
  // D = A(16x32 bf16) x B(32x16 bf16) + C(16x16 f32)
  return __builtin_amdgcn_wmma_f32_16x16x32_bf16(
      false, a, false, b, (short)0, c, false, false);
}

// A fragment 16x32 (MxK), source row-major [m][k]: lane half 'hi',
// VGPR i -> k = hi*16 + i*2
__device__ __forceinline__ v16bf load_a_frag(const unsigned short* p, int ld, int lane) {
  const int hi = (lane >> 4) & 1;
  const int m  = lane & 15;
  union { v16bf v; unsigned u[8]; } r;
  const unsigned short* row = p + m * ld;
#pragma unroll
  for (int i = 0; i < 8; ++i) {
    const int k = hi * 16 + i * 2;
    r.u[i] = *(const unsigned*)(row + k);
  }
  return r.v;
}

// B fragment 32x16 (KxN) from a source stored row-major [n][k]  (B = src^T)
__device__ __forceinline__ v16bf load_bT_frag(const unsigned short* p, int ld, int lane) {
  const int kh = ((lane >> 4) & 1) * 16;
  const int n  = lane & 15;
  union { v16bf v; unsigned u[8]; } r;
  const unsigned short* row = p + n * ld + kh;
#pragma unroll
  for (int i = 0; i < 8; ++i) r.u[i] = *(const unsigned*)(row + 2 * i);
  return r.v;
}

// B fragment 32x16 (KxN) from a source stored row-major [k][n]  (used for P*V)
__device__ __forceinline__ v16bf load_b_frag(const unsigned short* p, int ld, int lane) {
  const int kh = ((lane >> 4) & 1) * 16;
  const int n  = lane & 15;
  union { v16bf v; unsigned short s[16]; } r;
#pragma unroll
  for (int i = 0; i < 16; ++i) r.s[i] = p[(kh + i) * ld + n];
  return r.v;
}

// stage a 64x64 bf16 tile (row stride HD in memory) into LDS (stride 72)
__device__ __forceinline__ void copy_tile64(unsigned short (*dst)[72],
                                            const unsigned short* src, int tid) {
#if HAVE_ASYNC
#pragma unroll
  for (int i = 0; i < 4; ++i) {                 // 512 x b128 chunks / 128 thr
    const int e = i * 128 + tid, r = e >> 3, c8 = e & 7;
    ASYNC_COPY_B128(&dst[r][c8 * 8], src + r * HD + c8 * 8);
  }
#else
  const unsigned* s32 = (const unsigned*)src;
#pragma unroll
  for (int i = 0; i < 16; ++i) {                // 2048 x b32 / 128 thr
    const int e = i * 128 + tid, r = e >> 5, c = e & 31;
    *(unsigned*)&dst[r][c * 2] = s32[r * 32 + c];
  }
#endif
}

// ---------- kernel 1: QKV projection  C[8192,2304] = X * Wqkv^T --------------
// bf16 out, layout [3][B][H][N][64]; Q pre-scaled by 1/sqrt(hd). K-tile = 64.

__global__ __launch_bounds__(256) void qkv_proj_kernel(
    const float* __restrict__ x, const float* __restrict__ w,
    unsigned short* __restrict__ qkv) {
  __shared__ __align__(16) unsigned short Xs[128][68];
  __shared__ __align__(16) unsigned short Ws[64][68];

  const int tid = threadIdx.x, lane = tid & 31, wave = tid >> 5;
  const int wm = wave >> 1, wn = wave & 1;              // 4x2 wave grid
  const int hi = (lane >> 4) & 1, l15 = lane & 15;
  const int mblk = blockIdx.x, nblk = blockIdx.y;

  v8f acc[2][2];
#pragma unroll
  for (int a = 0; a < 2; ++a)
#pragma unroll
    for (int c = 0; c < 2; ++c) acc[a][c] = zero_v8();

  for (int k0 = 0; k0 < DIM; k0 += 64) {
#pragma unroll
    for (int i = 0; i < 8; ++i) {                       // X: 128x64 f32 -> bf16
      const int e = i * 256 + tid, r = e >> 4, c4 = e & 15;
      const float4 f = *(const float4*)&x[(size_t)(mblk * 128 + r) * DIM + k0 + c4 * 4];
      union { unsigned long long u; unsigned short s[4]; } pk;
      pk.s[0] = f2bf(f.x); pk.s[1] = f2bf(f.y);
      pk.s[2] = f2bf(f.z); pk.s[3] = f2bf(f.w);
      *(unsigned long long*)&Xs[r][c4 * 4] = pk.u;
    }
#pragma unroll
    for (int i = 0; i < 4; ++i) {                       // W: 64x64 f32 -> bf16
      const int e = i * 256 + tid, r = e >> 4, c4 = e & 15;
      const float4 f = *(const float4*)&w[(size_t)(nblk * 64 + r) * DIM + k0 + c4 * 4];
      union { unsigned long long u; unsigned short s[4]; } pk;
      pk.s[0] = f2bf(f.x); pk.s[1] = f2bf(f.y);
      pk.s[2] = f2bf(f.z); pk.s[3] = f2bf(f.w);
      *(unsigned long long*)&Ws[r][c4 * 4] = pk.u;
    }
    __syncthreads();

#pragma unroll
    for (int ks = 0; ks < 2; ++ks) {
      v16bf bf[2];
#pragma unroll
      for (int nf = 0; nf < 2; ++nf)
        bf[nf] = load_bT_frag(&Ws[wn * 32 + nf * 16][ks * 32], 68, lane);
#pragma unroll
      for (int mf = 0; mf < 2; ++mf) {
        v16bf af = load_a_frag(&Xs[wm * 32 + mf * 16][ks * 32], 68, lane);
#pragma unroll
        for (int nf = 0; nf < 2; ++nf)
          acc[mf][nf] = wmma_bf16(af, bf[nf], acc[mf][nf]);
      }
    }
    __syncthreads();
  }

  // 64-wide, 64-aligned N range lies inside one (qkv-split, head) bucket
  const int dbase = nblk * 64;
  const int s3 = dbase / DIM;
  const int hh = (dbase % DIM) >> 6;
  const float qs = (s3 == 0) ? SCALE_F : 1.0f;
#pragma unroll
  for (int mf = 0; mf < 2; ++mf)
#pragma unroll
    for (int r = 0; r < 8; ++r) {
      const int m = mblk * 128 + wm * 32 + mf * 16 + hi * 8 + r;
      const int bb = m >> 10, nsq = m & 1023;
#pragma unroll
      for (int nf = 0; nf < 2; ++nf) {
        const int e = wn * 32 + nf * 16 + l15;
        const size_t idx =
            ((((size_t)s3 * BATCH + bb) * HEADS + hh) * NSEQ + nsq) * HD + e;
        qkv[idx] = f2bf(acc[mf][nf][r] * qs);
      }
    }
}

// ---------- kernel 2: fused flash attention ----------------------------------
// grid = B*H*8 blocks of 128 threads (4 waves); wave owns 32 query rows.
// K/V tiles double-buffered in LDS via async b128 copies.

__global__ __launch_bounds__(128) void flash_attn_kernel(
    const unsigned short* __restrict__ qkv, unsigned short* __restrict__ o_bf) {
  __shared__ __align__(16) unsigned short Ks[2][64][72];
  __shared__ __align__(16) unsigned short Vs[2][64][72];
  __shared__ __align__(16) unsigned short Ps[4][32][68];   // per-wave P staging

  const int tid  = threadIdx.x;
  const int lane = tid & 31;
  const int wave = tid >> 5;
  const int hi   = (lane >> 4) & 1;
  const int l15  = lane & 15;

  const int bh   = blockIdx.x >> 3;
  const int qblk = blockIdx.x & 7;
  const int bb   = bh / HEADS;
  const int hh   = bh % HEADS;
  const int qrow0 = qblk * 128 + wave * 32;

  const size_t TS = (size_t)BATCH * HEADS * NSEQ * HD;
  const unsigned short* Qg = qkv + (size_t)bh * NSEQ * HD;
  const unsigned short* Kg = qkv + TS + (size_t)bh * NSEQ * HD;
  const unsigned short* Vg = qkv + 2 * TS + (size_t)bh * NSEQ * HD;

  // Q fragments stay in registers for the whole kernel (Q already scaled)
  v16bf qa[2][2];
#pragma unroll
  for (int mf = 0; mf < 2; ++mf)
#pragma unroll
    for (int ks = 0; ks < 2; ++ks)
      qa[mf][ks] = load_a_frag(Qg + (size_t)(qrow0 + mf * 16) * HD + ks * 32, HD, lane);

  v8f o[2][4];
  float mstat[2][8], lstat[2][8];
#pragma unroll
  for (int mf = 0; mf < 2; ++mf) {
#pragma unroll
    for (int nf = 0; nf < 4; ++nf) o[mf][nf] = zero_v8();
#pragma unroll
    for (int r = 0; r < 8; ++r) { mstat[mf][r] = -1e30f; lstat[mf][r] = 0.0f; }
  }

  // prologue: stage tile 0
  copy_tile64(Ks[0], Kg, tid);
  copy_tile64(Vs[0], Vg, tid);
  ASYNC_WAIT();
  __syncthreads();

  for (int kt = 0; kt < NSEQ / 64; ++kt) {
    const int cur = kt & 1;
    if (kt + 1 < NSEQ / 64) {       // prefetch next tile into other buffer
      copy_tile64(Ks[cur ^ 1], Kg + (size_t)(kt + 1) * 64 * HD, tid);
      copy_tile64(Vs[cur ^ 1], Vg + (size_t)(kt + 1) * 64 * HD, tid);
    }

    // S = Q * K^T  (32 rows x 64 keys per wave)
    v8f sfr[2][4];
#pragma unroll
    for (int mf = 0; mf < 2; ++mf)
#pragma unroll
      for (int nf = 0; nf < 4; ++nf) sfr[mf][nf] = zero_v8();
#pragma unroll
    for (int nf = 0; nf < 4; ++nf)
#pragma unroll
      for (int ks = 0; ks < 2; ++ks) {
        v16bf kf = load_bT_frag(&Ks[cur][nf * 16][ks * 32], 72, lane);
#pragma unroll
        for (int mf = 0; mf < 2; ++mf)
          sfr[mf][nf] = wmma_bf16(qa[mf][ks], kf, sfr[mf][nf]);
      }

    // online softmax; a row lives across the 16 lanes of a half (C layout)
#pragma unroll
    for (int mf = 0; mf < 2; ++mf) {
      float nm[8], sc[8];
#pragma unroll
      for (int r = 0; r < 8; ++r) {
        float t = fmaxf(fmaxf(sfr[mf][0][r], sfr[mf][1][r]),
                        fmaxf(sfr[mf][2][r], sfr[mf][3][r]));
        t = fmaxf(t, __shfl_xor(t, 1, 32));
        t = fmaxf(t, __shfl_xor(t, 2, 32));
        t = fmaxf(t, __shfl_xor(t, 4, 32));
        t = fmaxf(t, __shfl_xor(t, 8, 32));
        nm[r] = fmaxf(mstat[mf][r], t);
        sc[r] = __expf(mstat[mf][r] - nm[r]);
        mstat[mf][r] = nm[r];
      }
#pragma unroll
      for (int r = 0; r < 8; ++r) {
        float rs = 0.0f;
#pragma unroll
        for (int nf = 0; nf < 4; ++nf) {
          float p = __expf(sfr[mf][nf][r] - nm[r]);
          sfr[mf][nf][r] = p;
          rs += p;
        }
        rs += __shfl_xor(rs, 1, 32);
        rs += __shfl_xor(rs, 2, 32);
        rs += __shfl_xor(rs, 4, 32);
        rs += __shfl_xor(rs, 8, 32);
        lstat[mf][r] = lstat[mf][r] * sc[r] + rs;
      }
#pragma unroll
      for (int nf = 0; nf < 4; ++nf)
#pragma unroll
        for (int r = 0; r < 8; ++r) {
          o[mf][nf][r] *= sc[r];
          // C-layout -> LDS (wave-private); same-wave DS ops are in-order
          Ps[wave][mf * 16 + hi * 8 + r][nf * 16 + l15] = f2bf(sfr[mf][nf][r]);
        }
    }

    // O += P * V
    v16bf vb[2][4];
#pragma unroll
    for (int ks = 0; ks < 2; ++ks)
#pragma unroll
      for (int nf = 0; nf < 4; ++nf)
        vb[ks][nf] = load_b_frag(&Vs[cur][ks * 32][nf * 16], 72, lane);
#pragma unroll
    for (int mf = 0; mf < 2; ++mf)
#pragma unroll
      for (int ks = 0; ks < 2; ++ks) {
        v16bf pa = load_a_frag(&Ps[wave][mf * 16][ks * 32], 68, lane);
#pragma unroll
        for (int nf = 0; nf < 4; ++nf)
          o[mf][nf] = wmma_bf16(pa, vb[ks][nf], o[mf][nf]);
      }

    ASYNC_WAIT();                   // my prefetch writes are in LDS
    __syncthreads();                // everyone done with 'cur', prefetch visible
  }

  // normalize, store bf16 [B*N, DIM] (col = h*64 + d) for the out-proj GEMM
#pragma unroll
  for (int mf = 0; mf < 2; ++mf)
#pragma unroll
    for (int r = 0; r < 8; ++r) {
      const float inv = 1.0f / lstat[mf][r];
      const int m = qrow0 + mf * 16 + hi * 8 + r;
#pragma unroll
      for (int nf = 0; nf < 4; ++nf) {
        const int d = nf * 16 + l15;
        o_bf[((size_t)(bb * NSEQ + m)) * DIM + hh * HD + d] = f2bf(o[mf][nf][r] * inv);
      }
    }
}

// ---------- kernel 3: output projection  out = A * Wproj^T + bias ------------

__global__ __launch_bounds__(256) void out_proj_kernel(
    const unsigned short* __restrict__ abf, const float* __restrict__ w,
    const float* __restrict__ bias, float* __restrict__ out) {
  __shared__ __align__(16) unsigned short As[128][72];
  __shared__ __align__(16) unsigned short Ws[64][68];

  const int tid = threadIdx.x, lane = tid & 31, wave = tid >> 5;
  const int wm = wave >> 1, wn = wave & 1;
  const int hi = (lane >> 4) & 1, l15 = lane & 15;
  const int mblk = blockIdx.x, nblk = blockIdx.y;

  v8f acc[2][2];
#pragma unroll
  for (int a = 0; a < 2; ++a)
#pragma unroll
    for (int c = 0; c < 2; ++c) acc[a][c] = zero_v8();

  for (int k0 = 0; k0 < DIM; k0 += 64) {
#if HAVE_ASYNC
#pragma unroll
    for (int i = 0; i < 4; ++i) {                 // A: 1024 b128 chunks
      const int e = i * 256 + tid, r = e >> 3, c8 = e & 7;
      ASYNC_COPY_B128(&As[r][c8 * 8],
                      abf + (size_t)(mblk * 128 + r) * DIM + k0 + c8 * 8);
    }
#else
    const unsigned* au = (const unsigned*)abf;
#pragma unroll
    for (int i = 0; i < 16; ++i) {
      const int e = i * 256 + tid, r = e >> 5, c = e & 31;
      *(unsigned*)&As[r][c * 2] =
          au[(size_t)(mblk * 128 + r) * (DIM / 2) + (k0 >> 1) + c];
    }
#endif
#pragma unroll
    for (int i = 0; i < 4; ++i) {                 // W: 64x64 f32 -> bf16
      const int e = i * 256 + tid, r = e >> 4, c4 = e & 15;
      const float4 f = *(const float4*)&w[(size_t)(nblk * 64 + r) * DIM + k0 + c4 * 4];
      union { unsigned long long u; unsigned short s[4]; } pk;
      pk.s[0] = f2bf(f.x); pk.s[1] = f2bf(f.y);
      pk.s[2] = f2bf(f.z); pk.s[3] = f2bf(f.w);
      *(unsigned long long*)&Ws[r][c4 * 4] = pk.u;
    }
    ASYNC_WAIT();
    __syncthreads();

#pragma unroll
    for (int ks = 0; ks < 2; ++ks) {
      v16bf bf[2];
#pragma unroll
      for (int nf = 0; nf < 2; ++nf)
        bf[nf] = load_bT_frag(&Ws[wn * 32 + nf * 16][ks * 32], 68, lane);
#pragma unroll
      for (int mf = 0; mf < 2; ++mf) {
        v16bf af = load_a_frag(&As[wm * 32 + mf * 16][ks * 32], 72, lane);
#pragma unroll
        for (int nf = 0; nf < 2; ++nf)
          acc[mf][nf] = wmma_bf16(af, bf[nf], acc[mf][nf]);
      }
    }
    __syncthreads();
  }

#pragma unroll
  for (int nf = 0; nf < 2; ++nf) {
    const int n = nblk * 64 + wn * 32 + nf * 16 + l15;
    const float bn = bias[n];
#pragma unroll
    for (int mf = 0; mf < 2; ++mf)
#pragma unroll
      for (int r = 0; r < 8; ++r) {
        const int m = mblk * 128 + wm * 32 + mf * 16 + hi * 8 + r;
        out[(size_t)m * DIM + n] = acc[mf][nf][r] + bn;
      }
  }
}

// ---------- launch -----------------------------------------------------------

extern "C" void kernel_launch(void* const* d_in, const int* in_sizes, int n_in,
                              void* d_out, int out_size, void* d_ws, size_t ws_size,
                              hipStream_t stream) {
  (void)in_sizes; (void)n_in; (void)out_size; (void)ws_size;
  const float* x      = (const float*)d_in[0];
  const float* w_qkv  = (const float*)d_in[1];
  const float* w_proj = (const float*)d_in[2];
  const float* b_proj = (const float*)d_in[3];
  float* out = (float*)d_out;

  const size_t TS = (size_t)BATCH * HEADS * NSEQ * HD;   // 6,291,456 elements
  unsigned short* qkv_ws = (unsigned short*)d_ws;        // 3*TS bf16
  unsigned short* o_ws   = qkv_ws + 3 * TS;              // TS bf16

  qkv_proj_kernel<<<dim3(64, 36), 256, 0, stream>>>(x, w_qkv, qkv_ws);
  flash_attn_kernel<<<dim3(BATCH * HEADS * 8), 128, 0, stream>>>(qkv_ws, o_ws);
  out_proj_kernel<<<dim3(64, 12), 256, 0, stream>>>(o_ws, w_proj, b_proj, out);
}
